// NequIP_77618648973817
// MI455X (gfx1250) — matrix operand order repaired and checked by reading, not compiled, not run on hardware
//
#include <hip/hip_runtime.h>
#include <hip/hip_bf16.h>
#include <math.h>

#define NN 40000
#define NE 640000
#define NG 64

typedef __bf16 bf16_t;
typedef bf16_t v16bf __attribute__((ext_vector_type(16)));
typedef float  v8f   __attribute__((ext_vector_type(8)));
typedef int    v4i   __attribute__((ext_vector_type(4)));

#if defined(__gfx1250__) && __has_builtin(__builtin_amdgcn_global_load_async_to_lds_b128)
#define ASYNC_OK 1
#else
#define ASYNC_OK 0
#endif

// Copy 16 aligned bytes global -> LDS. Async (no VGPR transit) when available.
__device__ __forceinline__ void async_copy16(const float* g, float* l) {
#if ASYNC_OK
  __builtin_amdgcn_global_load_async_to_lds_b128(
      (__attribute__((address_space(1))) v4i*)g,
      (__attribute__((address_space(3))) v4i*)l, 0, 0);
#else
  *(float4*)l = *(const float4*)g;
#endif
}

__device__ __forceinline__ void async_wait_all() {
#if ASYNC_OK
#if __has_builtin(__builtin_amdgcn_s_wait_asynccnt)
  __builtin_amdgcn_s_wait_asynccnt(0);
#else
  asm volatile("s_wait_asynccnt 0" ::: "memory");
#endif
#endif
}

__device__ __forceinline__ v8f wmma_bf16(v16bf a, v16bf b, v8f c) {
  // D = A(16x32 bf16) * B(32x16 bf16) + C(16x16 f32)
  return __builtin_amdgcn_wmma_f32_16x16x32_bf16(false, a, false, b, (short)0, c,
                                                 false, false);
}

// A fragment (16x32, bf16) from an LDS f32 tile.
// CDNA5 16-bit A layout: lane m=lane&15, hi=lane>>4;
// half h -> k = kbase + (h&7) + 16*(h>>3) + 8*hi. Element = S[m*lda + off + k*stride].
__device__ __forceinline__ v16bf load_A_frag(const float* S, int lda, int kmax,
                                             int kbase, int stride, int off) {
  const int lane = threadIdx.x & 31;
  const int m = lane & 15;
  const int hi = lane >> 4;
  v16bf a;
#pragma unroll
  for (int h = 0; h < 16; ++h) {
    const int k = kbase + (h & 7) + 16 * (h >> 3) + 8 * hi;
    float v = (k < kmax) ? S[m * lda + off + k * stride] : 0.0f;
    a[h] = (bf16_t)v;
  }
  return a;
}

// Pre-packed B fragment: 32 lanes x 16 bf16, lane-major (32B contiguous per lane).
__device__ __forceinline__ v16bf load_B_packed(const bf16_t* __restrict__ PB, int fid) {
  const int lane = threadIdx.x & 31;
  return *(const v16bf*)(PB + (size_t)fid * 512 + lane * 16);
}

__device__ __forceinline__ float siluf(float x) {
  return x * __builtin_amdgcn_rcpf(1.0f + __expf(-x));
}
__device__ __forceinline__ float sigmf(float x) {
  return __builtin_amdgcn_rcpf(1.0f + __expf(-x));
}

// ---------------------------------------------------------------------------
// Pack all WMMA B fragments (bf16, CDNA5 B lane layout) into workspace.
// fid layout per layer b (base b*28): 0-3 Wr1[nt], 4-11 Wr2[nt*2+kb32],
// 12-13 W0[nt], 14 W1, 15 W2, 16-17 W10[nt], 18-19 W20[nt], 20 W01, 21 W02,
// 22-23 S0[nt], 24 Wg1, 25 Wg2, 26 S1, 27 S2.  fid 84-87: Wout1[nt*2+kb32].
__global__ __launch_bounds__(32) void nequip_pack_B(
    const float* __restrict__ W0, const float* __restrict__ W1, const float* __restrict__ W2,
    const float* __restrict__ W10, const float* __restrict__ W20,
    const float* __restrict__ W01, const float* __restrict__ W02,
    const float* __restrict__ Wr1, const float* __restrict__ Wr2,
    const float* __restrict__ S0, const float* __restrict__ S1, const float* __restrict__ S2,
    const float* __restrict__ Wg1, const float* __restrict__ Wg2,
    const float* __restrict__ Wout1, bf16_t* __restrict__ PB) {
  const int fid = blockIdx.x;
  const int lane = threadIdx.x;
  const float* B;
  int ldb, n0, kmax, nmax, kbase;
  if (fid < 84) {
    const int b = fid / 28;
    const int f = fid % 28;
    if (f < 4)       { B = Wr1 + b * 512;  ldb = 64; n0 = f * 16;        kmax = 8;  nmax = 64; kbase = 0; }
    else if (f < 12) { int t = f - 4; B = Wr2 + b * 64 * 56; ldb = 56; n0 = (t >> 1) * 16; kmax = 64; nmax = 56; kbase = (t & 1) * 32; }
    else if (f < 14) { B = W0 + b * 1024;  ldb = 32; n0 = (f - 12) * 16; kmax = 32; nmax = 32; kbase = 0; }
    else if (f == 14){ B = W1 + b * 256;   ldb = 16; n0 = 0;             kmax = 16; nmax = 16; kbase = 0; }
    else if (f == 15){ B = W2 + b * 64;    ldb = 8;  n0 = 0;             kmax = 8;  nmax = 8;  kbase = 0; }
    else if (f < 18) { B = W10 + b * 512;  ldb = 32; n0 = (f - 16) * 16; kmax = 16; nmax = 32; kbase = 0; }
    else if (f < 20) { B = W20 + b * 256;  ldb = 32; n0 = (f - 18) * 16; kmax = 8;  nmax = 32; kbase = 0; }
    else if (f == 20){ B = W01 + b * 512;  ldb = 16; n0 = 0;             kmax = 32; nmax = 16; kbase = 0; }
    else if (f == 21){ B = W02 + b * 256;  ldb = 8;  n0 = 0;             kmax = 32; nmax = 8;  kbase = 0; }
    else if (f < 24) { B = S0 + b * 1024;  ldb = 32; n0 = (f - 22) * 16; kmax = 32; nmax = 32; kbase = 0; }
    else if (f == 24){ B = Wg1 + b * 512;  ldb = 16; n0 = 0;             kmax = 32; nmax = 16; kbase = 0; }
    else if (f == 25){ B = Wg2 + b * 256;  ldb = 8;  n0 = 0;             kmax = 32; nmax = 8;  kbase = 0; }
    else if (f == 26){ B = S1 + b * 256;   ldb = 16; n0 = 0;             kmax = 16; nmax = 16; kbase = 0; }
    else             { B = S2 + b * 64;    ldb = 8;  n0 = 0;             kmax = 8;  nmax = 8;  kbase = 0; }
  } else {
    const int t = fid - 84;
    B = Wout1; ldb = 32; n0 = (t >> 1) * 16; kmax = 56; nmax = 32; kbase = (t & 1) * 32;
  }
  const int n = n0 + (lane & 15);
  const int hi = lane >> 4;
  bf16_t* out = PB + (size_t)fid * 512 + lane * 16;
#pragma unroll
  for (int h = 0; h < 16; ++h) {
    const int k = kbase + (h & 7) + 16 * (h >> 3) + 8 * hi;
    float v = (k < kmax && n < nmax) ? B[k * ldb + n] : 0.0f;
    out[h] = (bf16_t)v;
  }
}

// ---------------------------------------------------------------------------
// x0 = embed0[z]
__global__ __launch_bounds__(256) void nequip_init_x0(const int* __restrict__ z,
                                                      const float* __restrict__ embed0,
                                                      float* __restrict__ X0) {
  int idx = blockIdx.x * 256 + threadIdx.x;
  if (idx >= NN * 32) return;
  X0[idx] = embed0[z[idx >> 5] * 32 + (idx & 31)];
}

// ---------------------------------------------------------------------------
// Per-edge geometry: rbf(8), env, Y1(3), Y2(5)
__global__ __launch_bounds__(256) void nequip_edge_geom(const float* __restrict__ pos,
                                                        const int* __restrict__ eidx,
                                                        float* __restrict__ RBF,
                                                        float* __restrict__ ENVB,
                                                        float* __restrict__ Y1B,
                                                        float* __restrict__ Y2B) {
  int e = blockIdx.x * 256 + threadIdx.x;
  if (e >= NE) return;
  int s = eidx[e];
  int t = eidx[NE + e];
  float rx = pos[t * 3 + 0] - pos[s * 3 + 0];
  float ry = pos[t * 3 + 1] - pos[s * 3 + 1];
  float rz = pos[t * 3 + 2] - pos[s * 3 + 2];
  float d = sqrtf(rx * rx + ry * ry + rz * rz + 1e-12f);
  float inv = __builtin_amdgcn_rcpf(d);
  float x = rx * inv, y = ry * inv, zc = rz * inv;
#pragma unroll
  for (int k = 0; k < 8; ++k) {
    float mu = (float)k * (5.0f / 7.0f);
    float dd = d - mu;
    RBF[e * 8 + k] = __expf(-dd * dd * (1.0f / (2.0f * 0.625f * 0.625f)));
  }
  float dm = fminf(d, 5.0f);
  ENVB[e] = 0.5f * (__cosf(3.14159265358979f * dm / 5.0f) + 1.0f);
  Y1B[e * 3 + 0] = x;
  Y1B[e * 3 + 1] = y;
  Y1B[e * 3 + 2] = zc;
  float c3 = 1.7320508075688772f;
  Y2B[e * 5 + 0] = c3 * x * y;
  Y2B[e * 5 + 1] = c3 * y * zc;
  Y2B[e * 5 + 2] = 0.5f * (3.0f * zc * zc - 1.0f);
  Y2B[e * 5 + 3] = c3 * zc * x;
  Y2B[e * 5 + 4] = 0.5f * c3 * (x * x - y * y);
}

// ---------------------------------------------------------------------------
// Edge message kernel: one wave32 per 16-edge tile, 2 waves per block.
__global__ __launch_bounds__(64) void nequip_edge_message(
    const float* __restrict__ X0, const float* __restrict__ X1, const float* __restrict__ X2,
    float* __restrict__ A0, float* __restrict__ A1, float* __restrict__ A2,
    const float* __restrict__ RBF, const float* __restrict__ ENVB,
    const float* __restrict__ Y1B, const float* __restrict__ Y2B,
    const int* __restrict__ eidx, const bf16_t* __restrict__ PB, int fb) {
  __shared__ float sA[2][16 * 64];
  __shared__ float sH[2][16 * 64];
  __shared__ float sx0[2][16 * 32];
  __shared__ float sx1[2][16 * 48];
  __shared__ float sx2[2][16 * 40];
  __shared__ float sy1[2][16 * 3];
  __shared__ float sy2[2][16 * 5];
  __shared__ float senv[2][16];
  __shared__ int ssrc[2][16];
  __shared__ int sdst[2][16];

  const int wave = threadIdx.x >> 5;
  const int lane = threadIdx.x & 31;
  const int hi = lane >> 4;
  const int nn = lane & 15;
  const int e0 = (blockIdx.x * 2 + wave) * 16;

  float* pA = sA[wave];
  float* pH = sH[wave];
  float* px0 = sx0[wave];
  float* px1 = sx1[wave];
  float* px2 = sx2[wave];
  float* py1 = sy1[wave];
  float* py2 = sy2[wave];
  float* penv = senv[wave];
  int* psrc = ssrc[wave];
  int* pdst = sdst[wave];

  if (lane < 16) {
    int e = e0 + lane;
    psrc[lane] = eidx[e];
    pdst[lane] = eidx[NE + e];
    penv[lane] = ENVB[e];
#pragma unroll
    for (int k = 0; k < 8; ++k) pA[lane * 64 + k] = RBF[e * 8 + k];
#pragma unroll
    for (int i = 0; i < 3; ++i) py1[lane * 3 + i] = Y1B[e * 3 + i];
#pragma unroll
    for (int i = 0; i < 5; ++i) py2[lane * 5 + i] = Y2B[e * 5 + i];
  }
  __syncthreads();

  // gather node features for src nodes: 16B chunks straight into LDS
#pragma unroll
  for (int t = 0; t < 4; ++t) {  // x0: 16 rows x 8 chunks
    int ch = lane + 32 * t;
    int r = ch >> 3, sub = ch & 7;
    async_copy16(X0 + (size_t)psrc[r] * 32 + sub * 4, px0 + r * 32 + sub * 4);
  }
#pragma unroll
  for (int t = 0; t < 6; ++t) {  // x1: 16 rows x 12 chunks
    int ch = lane + 32 * t;
    int r = ch / 12, sub = ch % 12;
    async_copy16(X1 + (size_t)psrc[r] * 48 + sub * 4, px1 + r * 48 + sub * 4);
  }
#pragma unroll
  for (int t = 0; t < 5; ++t) {  // x2: 16 rows x 10 chunks
    int ch = lane + 32 * t;
    int r = ch / 10, sub = ch % 10;
    async_copy16(X2 + (size_t)psrc[r] * 40 + sub * 4, px2 + r * 40 + sub * 4);
  }
  async_wait_all();
  __syncthreads();

  const v8f vzero = {0.f, 0.f, 0.f, 0.f, 0.f, 0.f, 0.f, 0.f};

  // h = silu(rbf @ Wr1) : (16x8)@(8x64)
  {
    v16bf a_rbf = load_A_frag(pA, 64, 8, 0, 1, 0);
#pragma unroll
    for (int nt = 0; nt < 4; ++nt) {
      v8f c = wmma_bf16(a_rbf, load_B_packed(PB, fb + nt), vzero);
#pragma unroll
      for (int j = 0; j < 8; ++j)
        pH[(j + 8 * hi) * 64 + nt * 16 + nn] = siluf(c[j]);
    }
  }
  __syncthreads();

  // w = (h @ Wr2) * env : (16x64)@(64x56)
  v8f wt[4];
#pragma unroll
  for (int nt = 0; nt < 4; ++nt) {
    v8f c = vzero;
#pragma unroll
    for (int kb = 0; kb < 2; ++kb) {
      v16bf a = load_A_frag(pH, 64, 64, kb * 32, 1, 0);
      c = wmma_bf16(a, load_B_packed(PB, fb + 4 + nt * 2 + kb), c);
    }
#pragma unroll
    for (int j = 0; j < 8; ++j) c[j] *= penv[j + 8 * hi];
    wt[nt] = c;
  }

  // x0s = x0[src] @ W0
  v8f x0s[2];
  {
    v16bf a = load_A_frag(px0, 32, 32, 0, 1, 0);
#pragma unroll
    for (int nt = 0; nt < 2; ++nt)
      x0s[nt] = wmma_bf16(a, load_B_packed(PB, fb + 12 + nt), vzero);
  }
  // x1s[:, :, i] = x1[src][:, :, i] @ W1
  v8f x1s[3];
#pragma unroll
  for (int i = 0; i < 3; ++i) {
    v16bf a = load_A_frag(px1, 48, 16, 0, 3, i);
    x1s[i] = wmma_bf16(a, load_B_packed(PB, fb + 14), vzero);
  }
  // x2s[:, :, m] = x2[src][:, :, m] @ W2
  v8f x2s[5];
#pragma unroll
  for (int i = 0; i < 5; ++i) {
    v16bf a = load_A_frag(px2, 40, 8, 0, 5, i);
    x2s[i] = wmma_bf16(a, load_B_packed(PB, fb + 15), vzero);
  }

  // dot1 = x1s . Y1, dot2 = x2s . Y2 (C layout) -> stage for A reuse
  float dot1v[8], dot2v[8];
#pragma unroll
  for (int j = 0; j < 8; ++j) {
    int e = j + 8 * hi;
    dot1v[j] = x1s[0][j] * py1[e * 3 + 0] + x1s[1][j] * py1[e * 3 + 1] +
               x1s[2][j] * py1[e * 3 + 2];
    float d2 = 0.0f;
#pragma unroll
    for (int i = 0; i < 5; ++i) d2 += x2s[i][j] * py2[e * 5 + i];
    dot2v[j] = d2;
  }
  __syncthreads();
#pragma unroll
  for (int j = 0; j < 8; ++j) {
    pA[(j + 8 * hi) * 64 + nn] = dot1v[j];
    pA[(j + 8 * hi) * 64 + 16 + nn] = dot2v[j];
    pH[(j + 8 * hi) * 64 + nn] = x0s[0][j];
    pH[(j + 8 * hi) * 64 + 16 + nn] = x0s[1][j];
  }
  __syncthreads();

  // m0 = (x0s + dot1@W10 + dot2@W20) * w0
  v8f m0[2];
  {
    v16bf a_d1 = load_A_frag(pA, 64, 16, 0, 1, 0);
    v16bf a_d2 = load_A_frag(pA, 64, 8, 0, 1, 16);
#pragma unroll
    for (int nt = 0; nt < 2; ++nt) {
      v8f c = x0s[nt];
      c = wmma_bf16(a_d1, load_B_packed(PB, fb + 16 + nt), c);
      c = wmma_bf16(a_d2, load_B_packed(PB, fb + 18 + nt), c);
#pragma unroll
      for (int j = 0; j < 8; ++j) c[j] *= wt[nt][j];
      m0[nt] = c;
    }
  }
  // u1 = x0s @ W01, u2 = x0s @ W02
  v8f u1, u2;
  {
    v16bf a = load_A_frag(pH, 64, 32, 0, 1, 0);
    u1 = wmma_bf16(a, load_B_packed(PB, fb + 20), vzero);
    u2 = wmma_bf16(a, load_B_packed(PB, fb + 21), vzero);
  }

  // scatter
#pragma unroll
  for (int nt = 0; nt < 2; ++nt)
#pragma unroll
    for (int j = 0; j < 8; ++j) {
      int e = j + 8 * hi;
      atomicAdd(&A0[pdst[e] * 32 + nt * 16 + nn], m0[nt][j]);
    }
#pragma unroll
  for (int i = 0; i < 3; ++i)
#pragma unroll
    for (int j = 0; j < 8; ++j) {
      int e = j + 8 * hi;
      float v = (u1[j] * py1[e * 3 + i] + x1s[i][j]) * wt[2][j];
      atomicAdd(&A1[(pdst[e] * 16 + nn) * 3 + i], v);
    }
  if (nn < 8) {
#pragma unroll
    for (int i = 0; i < 5; ++i)
#pragma unroll
      for (int j = 0; j < 8; ++j) {
        int e = j + 8 * hi;
        float v = (u2[j] * py2[e * 5 + i] + x2s[i][j]) * wt[3][j];
        atomicAdd(&A2[(pdst[e] * 8 + nn) * 5 + i], v);
      }
  }
}

// ---------------------------------------------------------------------------
// Node update: one wave32 per 16-node tile, 4 waves per block.
__global__ __launch_bounds__(128) void nequip_node_update(
    float* __restrict__ X0, float* __restrict__ X1, float* __restrict__ X2,
    const float* __restrict__ A0, const float* __restrict__ A1, const float* __restrict__ A2,
    const bf16_t* __restrict__ PB, int fb) {
  __shared__ float sA[4][16 * 32];
  __shared__ float sB1[4][16 * 48];
  __shared__ float sB2[4][16 * 40];
  const int wave = threadIdx.x >> 5;
  const int lane = threadIdx.x & 31;
  const int hi = lane >> 4;
  const int nn = lane & 15;
  const int n0 = (blockIdx.x * 4 + wave) * 16;
  float* pA = sA[wave];
  float* pB1 = sB1[wave];
  float* pB2 = sB2[wave];

#pragma unroll
  for (int t = 0; t < 4; ++t) {
    int ch = lane + 32 * t;
    async_copy16(A0 + (size_t)n0 * 32 + ch * 4, pA + ch * 4);
  }
#pragma unroll
  for (int t = 0; t < 6; ++t) {
    int ch = lane + 32 * t;
    async_copy16(A1 + (size_t)n0 * 48 + ch * 4, pB1 + ch * 4);
  }
#pragma unroll
  for (int t = 0; t < 5; ++t) {
    int ch = lane + 32 * t;
    async_copy16(A2 + (size_t)n0 * 40 + ch * 4, pB2 + ch * 4);
  }
  async_wait_all();
  __syncthreads();

  const v8f vzero = {0.f, 0.f, 0.f, 0.f, 0.f, 0.f, 0.f, 0.f};
  v16bf a_a0 = load_A_frag(pA, 32, 32, 0, 1, 0);

  // x0 = silu(x0 + a0 @ S0)
#pragma unroll
  for (int nt = 0; nt < 2; ++nt) {
    v8f c = wmma_bf16(a_a0, load_B_packed(PB, fb + 22 + nt), vzero);
#pragma unroll
    for (int j = 0; j < 8; ++j) {
      int n = n0 + j + 8 * hi;
      int col = nt * 16 + nn;
      float v = X0[n * 32 + col] + c[j];
      X0[n * 32 + col] = siluf(v);
    }
  }
  v8f g1 = wmma_bf16(a_a0, load_B_packed(PB, fb + 24), vzero);
  v8f g2 = wmma_bf16(a_a0, load_B_packed(PB, fb + 25), vzero);
#pragma unroll
  for (int j = 0; j < 8; ++j) {
    g1[j] = sigmf(g1[j]);
    g2[j] = sigmf(g2[j]);
  }
  // x1 += (a1 @ S1) * g1
#pragma unroll
  for (int i = 0; i < 3; ++i) {
    v16bf a = load_A_frag(pB1, 48, 16, 0, 3, i);
    v8f c = wmma_bf16(a, load_B_packed(PB, fb + 26), vzero);
#pragma unroll
    for (int j = 0; j < 8; ++j) {
      int n = n0 + j + 8 * hi;
      X1[(n * 16 + nn) * 3 + i] += c[j] * g1[j];
    }
  }
  // x2 += (a2 @ S2) * g2
#pragma unroll
  for (int i = 0; i < 5; ++i) {
    v16bf a = load_A_frag(pB2, 40, 8, 0, 5, i);
    v8f c = wmma_bf16(a, load_B_packed(PB, fb + 27), vzero);
    if (nn < 8) {
#pragma unroll
      for (int j = 0; j < 8; ++j) {
        int n = n0 + j + 8 * hi;
        X2[(n * 8 + nn) * 5 + i] += c[j] * g2[j];
      }
    }
  }
}

// ---------------------------------------------------------------------------
// Readout
__global__ __launch_bounds__(128) void nequip_readout(
    const float* __restrict__ X0, const float* __restrict__ X1, const float* __restrict__ X2,
    const int* __restrict__ z, const int* __restrict__ batch,
    const bf16_t* __restrict__ PB, const float* __restrict__ Wout2,
    const float* __restrict__ shift, float* __restrict__ OUT) {
  __shared__ float sF[4][16 * 64];
  __shared__ float sHh[4][16 * 32];
  const int wave = threadIdx.x >> 5;
  const int lane = threadIdx.x & 31;
  const int hi = lane >> 4;
  const int nn = lane & 15;
  const int n0 = (blockIdx.x * 4 + wave) * 16;
  float* pF = sF[wave];
  float* pH = sHh[wave];

#pragma unroll
  for (int t = 0; t < 4; ++t) {  // x0 -> feat cols 0..31
    int ch = lane + 32 * t;
    int r = ch >> 3, sub = ch & 7;
    async_copy16(X0 + (size_t)(n0 + r) * 32 + sub * 4, pF + r * 64 + sub * 4);
  }
  for (int idx = lane; idx < 16 * 16; idx += 32) {
    int r = idx >> 4, c = idx & 15;
    float s = 1e-12f;
#pragma unroll
    for (int i = 0; i < 3; ++i) {
      float v = X1[((n0 + r) * 16 + c) * 3 + i];
      s += v * v;
    }
    pF[r * 64 + 32 + c] = sqrtf(s);
  }
  for (int idx = lane; idx < 16 * 8; idx += 32) {
    int r = idx >> 3, c = idx & 7;
    float s = 1e-12f;
#pragma unroll
    for (int i = 0; i < 5; ++i) {
      float v = X2[((n0 + r) * 8 + c) * 5 + i];
      s += v * v;
    }
    pF[r * 64 + 48 + c] = sqrtf(s);
  }
  async_wait_all();
  __syncthreads();

  const v8f vzero = {0.f, 0.f, 0.f, 0.f, 0.f, 0.f, 0.f, 0.f};
#pragma unroll
  for (int nt = 0; nt < 2; ++nt) {
    v8f c = vzero;
#pragma unroll
    for (int kb = 0; kb < 2; ++kb) {
      v16bf a = load_A_frag(pF, 64, 56, kb * 32, 1, 0);
      c = wmma_bf16(a, load_B_packed(PB, 84 + nt * 2 + kb), c);
    }
#pragma unroll
    for (int j = 0; j < 8; ++j)
      pH[(j + 8 * hi) * 32 + nt * 16 + nn] = siluf(c[j]);
  }
  __syncthreads();

  if (lane < 16) {
    int n = n0 + lane;
    float acc = 0.0f;
#pragma unroll
    for (int c = 0; c < 32; ++c) acc += pH[lane * 32 + c] * Wout2[c];
    acc += shift[z[n]];
    atomicAdd(&OUT[batch[n]], acc);
  }
}

// ---------------------------------------------------------------------------
extern "C" void kernel_launch(void* const* d_in, const int* in_sizes, int n_in,
                              void* d_out, int out_size, void* d_ws, size_t ws_size,
                              hipStream_t stream) {
  const int* z = (const int*)d_in[0];
  const float* pos = (const float*)d_in[1];
  const int* eidx = (const int*)d_in[2];
  const int* batch = (const int*)d_in[3];
  const float* embed0 = (const float*)d_in[4];
  const float* W0 = (const float*)d_in[5];
  const float* W1 = (const float*)d_in[6];
  const float* W2 = (const float*)d_in[7];
  const float* W10 = (const float*)d_in[8];
  const float* W20 = (const float*)d_in[9];
  const float* W01 = (const float*)d_in[10];
  const float* W02 = (const float*)d_in[11];
  const float* Wr1 = (const float*)d_in[12];
  const float* Wr2 = (const float*)d_in[13];
  const float* S0 = (const float*)d_in[14];
  const float* S1 = (const float*)d_in[15];
  const float* S2 = (const float*)d_in[16];
  const float* Wg1 = (const float*)d_in[17];
  const float* Wg2 = (const float*)d_in[18];
  const float* Wout1 = (const float*)d_in[19];
  const float* Wout2 = (const float*)d_in[20];
  const float* shift = (const float*)d_in[21];

  float* ws = (float*)d_ws;
  float* X0 = ws;
  float* X1 = X0 + (size_t)NN * 32;
  float* X2 = X1 + (size_t)NN * 48;
  float* A0 = X2 + (size_t)NN * 40;
  float* A1 = A0 + (size_t)NN * 32;
  float* A2 = A1 + (size_t)NN * 48;
  float* RBF = A2 + (size_t)NN * 40;
  float* ENVB = RBF + (size_t)NE * 8;
  float* Y1B = ENVB + (size_t)NE;
  float* Y2B = Y1B + (size_t)NE * 3;
  bf16_t* PB = (bf16_t*)(Y2B + (size_t)NE * 5);  // 88 fragments * 512 bf16

  nequip_pack_B<<<88, 32, 0, stream>>>(W0, W1, W2, W10, W20, W01, W02, Wr1, Wr2,
                                       S0, S1, S2, Wg1, Wg2, Wout1, PB);
  (void)hipMemsetAsync(X1, 0, (size_t)NN * 88 * sizeof(float), stream);
  nequip_init_x0<<<(NN * 32) / 256, 256, 0, stream>>>(z, embed0, X0);
  nequip_edge_geom<<<NE / 256, 256, 0, stream>>>(pos, eidx, RBF, ENVB, Y1B, Y2B);

  for (int b = 0; b < 3; ++b) {
    (void)hipMemsetAsync(A0, 0, (size_t)NN * 120 * sizeof(float), stream);
    nequip_edge_message<<<(NE / 16) / 2, 64, 0, stream>>>(
        X0, X1, X2, A0, A1, A2, RBF, ENVB, Y1B, Y2B, eidx, PB, b * 28);
    nequip_node_update<<<(NN / 16) / 4, 128, 0, stream>>>(
        X0, X1, X2, A0, A1, A2, PB, b * 28);
  }

  (void)hipMemsetAsync(d_out, 0, (size_t)NG * sizeof(float), stream);
  nequip_readout<<<(NN / 16) / 4, 128, 0, stream>>>(
      X0, X1, X2, z, batch, PB, Wout2, shift, (float*)d_out);
}
